// MatryoshkaTransformer_24300924960803
// MI455X (gfx1250) — matrix operand order repaired and checked by reading, not compiled, not run on hardware
//
#include <hip/hip_runtime.h>
#include <math.h>

// ---------------------------------------------------------------------------
// Types for CDNA5 WMMA (wave32): v_wmma_f32_16x16x32_f16
// A: 16x32 f16 (v16h), B: 32x16 f16 (v16h), C/D: 16x16 f32 (v8f)
// ---------------------------------------------------------------------------
typedef _Float16 half8 __attribute__((ext_vector_type(8)));
typedef _Float16 v16h  __attribute__((ext_vector_type(16)));
typedef float    v8f   __attribute__((ext_vector_type(8)));

#define WMMA_F16(A, B, C) \
  __builtin_amdgcn_wmma_f32_16x16x32_f16(false, (A), false, (B), (short)0, (C), false, false)

// ---------------------------------------------------------------------------
// Model constants
// ---------------------------------------------------------------------------
#define MTOK   8192        // B*T = 8*1024
#define DMODEL 1024
#define TSEQ   1024
#define NBATCH 8
#define NHEADS 16          // 4 (tier0) + 12 (tier1), all d_k = 64
#define DFF    4096        // FF = [1024, 3072] concatenated

// ---------------------------------------------------------------------------
// Assemble transposed f16 weight  dst[N][K]  from block-upper-triangular f32
// blocks (b00:[k0,n0], b01:[k0,N-n0], b11:[K-k0,N-n0]); the (k>=k0, n<n0)
// block is zero, or the folded low-rank term sum_r fkv[k-k0, h*8+r]*pkv[h,r,d]
// (used to fold the tier-0 K/V augmentation into Wk/Wv).
// ---------------------------------------------------------------------------
__global__ __launch_bounds__(256)
void build_wt(_Float16* __restrict__ dst,
              const float* __restrict__ b00, const float* __restrict__ b01,
              const float* __restrict__ b11,
              const float* __restrict__ fkv, const float* __restrict__ pkv,
              int N, int K, int n0, int k0)
{
  int idx = blockIdx.x * 256 + threadIdx.x;
  if (idx >= N * K) return;
  int k = idx % K;
  int n = idx / K;
  float v;
  if (k < k0) {
    v = (n < n0) ? b00[(size_t)k * n0 + n]
                 : b01[(size_t)k * (N - n0) + (n - n0)];
  } else if (n >= n0) {
    v = b11[(size_t)(k - k0) * (N - n0) + (n - n0)];
  } else if (fkv != nullptr) {
    int hh = n >> 6, dd = n & 63;
    v = 0.f;
#pragma unroll
    for (int r = 0; r < 8; ++r)
      v += fkv[(size_t)(k - k0) * 32 + hh * 8 + r] * pkv[(size_t)(hh * 8 + r) * 64 + dd];
  } else {
    v = 0.f;
  }
  dst[(size_t)n * K + k] = (_Float16)v;
}

// ---------------------------------------------------------------------------
// Per-token block layernorm: segments [0,256) and [256,1024) normalized
// independently (mean/var per segment), scaled by g,b; output f16.
// One 256-thread block per token.
// ---------------------------------------------------------------------------
__global__ __launch_bounds__(256)
void block_norm(const float* __restrict__ x,
                const float* __restrict__ g0, const float* __restrict__ g1,
                const float* __restrict__ b0, const float* __restrict__ b1,
                _Float16* __restrict__ out)
{
  __shared__ float s1[256];
  __shared__ float s2[256];
  const int tok = blockIdx.x, tid = threadIdx.x;
  const float* row = x + (size_t)tok * DMODEL;
  _Float16* orow = out + (size_t)tok * DMODEL;

  // segment 0: 256 elements, one per thread
  float v0 = row[tid];
  s1[tid] = v0; s2[tid] = v0 * v0;
  __syncthreads();
  for (int off = 128; off > 0; off >>= 1) {
    if (tid < off) { s1[tid] += s1[tid + off]; s2[tid] += s2[tid + off]; }
    __syncthreads();
  }
  float mu = s1[0] * (1.f / 256.f);
  float var = s2[0] * (1.f / 256.f) - mu * mu;
  float rs = rsqrtf(var + 1e-5f);
  orow[tid] = (_Float16)((v0 - mu) * rs * g0[tid] + b0[tid]);
  __syncthreads();

  // segment 1: 768 elements, three per thread (coalesced)
  float a = row[256 + tid], bb = row[512 + tid], c = row[768 + tid];
  s1[tid] = a + bb + c; s2[tid] = a * a + bb * bb + c * c;
  __syncthreads();
  for (int off = 128; off > 0; off >>= 1) {
    if (tid < off) { s1[tid] += s1[tid + off]; s2[tid] += s2[tid + off]; }
    __syncthreads();
  }
  mu = s1[0] * (1.f / 768.f);
  var = s2[0] * (1.f / 768.f) - mu * mu;
  rs = rsqrtf(var + 1e-5f);
  orow[256 + tid] = (_Float16)((a  - mu) * rs * g1[tid]       + b1[tid]);
  orow[512 + tid] = (_Float16)((bb - mu) * rs * g1[256 + tid] + b1[256 + tid]);
  orow[768 + tid] = (_Float16)((c  - mu) * rs * g1[512 + tid] + b1[512 + tid]);
}

// ---------------------------------------------------------------------------
// WMMA GEMM:  C[M,N] = A[M,K](f16,row-major, lda) * B (given as BT[N][K], f16)
// 256 threads = 8 waves (2x4); block tile 128x256; wave tile 64x64 (16 WMMA
// per K-step); K-step 32; register-staged double-buffered LDS so the next
// K-slab's global loads overlap the current slab's WMMAs.
// MODE 0: store f16 (ldc)     MODE 1: exact GELU then store f16
// MODE 2: f32 accumulate into Cacc (residual stream)
// Fragment layouts per CDNA5 ISA:
//   A frag (lane m=lane&15, hi=lane>>4): halfs 0..7  <- K = hi*8   .. +8
//                                        halfs 8..15 <- K = 16+hi*8.. +8
//   B frag (lane n=lane&15, hi):         same K order as A frag
//   D (r=0..7): row = r + 8*hi, col = lane&15
// ---------------------------------------------------------------------------
template <int MODE>
__global__ __launch_bounds__(256)
void gemm_wmma(const _Float16* __restrict__ A, const _Float16* __restrict__ BT,
               _Float16* __restrict__ C16, float* __restrict__ Cacc,
               int K, int lda, int ldc)
{
  __shared__ _Float16 As[2][128 * 40];   // padded stride 40 halfs (80B, 16B-aligned)
  __shared__ _Float16 Bs[2][256 * 40];
  const int tid  = threadIdx.x;
  const int lane = tid & 31;
  const int wave = tid >> 5;
  const int wm = wave >> 2;            // 0..1  (64-row slab)
  const int wn = wave & 3;             // 0..3  (64-col slab)
  const int lm = lane & 15, lhi = lane >> 4;
  const int m0 = blockIdx.x * 128;
  const int n0 = blockIdx.y * 256;

  // staging map: A: thread -> row tid>>1, 16-half seg; B: thread -> row tid, 32 halfs
  const int arow = tid >> 1;
  const int aseg = (tid & 1) << 4;
  const _Float16* Ag = A  + (size_t)(m0 + arow) * lda + aseg;
  const _Float16* Bg = BT + (size_t)(n0 + tid) * K;
  _Float16* Asw = &As[0][arow * 40 + aseg];
  _Float16* Bsw = &Bs[0][tid * 40];
  const int ABUF = 128 * 40;           // halfs per A buffer
  const int BBUF = 256 * 40;

  // prologue: fetch k=0 slab and stage it
  half8 ra0 = *(const half8*)(Ag);
  half8 ra1 = *(const half8*)(Ag + 8);
  half8 rb0 = *(const half8*)(Bg);
  half8 rb1 = *(const half8*)(Bg + 8);
  half8 rb2 = *(const half8*)(Bg + 16);
  half8 rb3 = *(const half8*)(Bg + 24);
  *(half8*)(Asw)      = ra0;
  *(half8*)(Asw + 8)  = ra1;
  *(half8*)(Bsw)      = rb0;
  *(half8*)(Bsw + 8)  = rb1;
  *(half8*)(Bsw + 16) = rb2;
  *(half8*)(Bsw + 24) = rb3;
  __syncthreads();

  v8f acc[4][4] = {};
  int buf = 0;

  for (int k0 = 0; k0 < K; k0 += 32) {
    const bool more = (k0 + 32 < K);
    // issue next slab's global loads; latency hidden under the WMMAs below
    if (more) {
      ra0 = *(const half8*)(Ag + k0 + 32);
      ra1 = *(const half8*)(Ag + k0 + 40);
      rb0 = *(const half8*)(Bg + k0 + 32);
      rb1 = *(const half8*)(Bg + k0 + 40);
      rb2 = *(const half8*)(Bg + k0 + 48);
      rb3 = *(const half8*)(Bg + k0 + 56);
      if (k0 + 64 < K) {
        __builtin_prefetch(Ag + k0 + 64, 0, 0);   // global_prefetch_b8
        __builtin_prefetch(Bg + k0 + 64, 0, 0);
      }
    }

    // compute current slab: 4 A-frags, 4 B-frags, 16 WMMAs
    const _Float16* Ab = &As[buf][0];
    const _Float16* Bb = &Bs[buf][0];
    v16h af[4], bf[4];
#pragma unroll
    for (int mt = 0; mt < 4; ++mt) {
      const _Float16* p = Ab + (wm * 64 + mt * 16 + lm) * 40 + lhi * 8;
      half8 c0 = *(const half8*)p;
      half8 c1 = *(const half8*)(p + 16);
#pragma unroll
      for (int i = 0; i < 8; ++i) { af[mt][i] = c0[i]; af[mt][i + 8] = c1[i]; }
    }
#pragma unroll
    for (int nt = 0; nt < 4; ++nt) {
      const _Float16* p = Bb + (wn * 64 + nt * 16 + lm) * 40 + lhi * 8;
      half8 c0 = *(const half8*)p;
      half8 c1 = *(const half8*)(p + 16);
#pragma unroll
      for (int i = 0; i < 8; ++i) { bf[nt][i] = c0[i]; bf[nt][i + 8] = c1[i]; }
    }
#pragma unroll
    for (int mt = 0; mt < 4; ++mt)
#pragma unroll
      for (int nt = 0; nt < 4; ++nt)
        acc[mt][nt] = WMMA_F16(af[mt], bf[nt], acc[mt][nt]);

    __syncthreads();
    if (more) {
      _Float16* aw = &As[0][0] + (buf ^ 1) * ABUF + arow * 40 + aseg;
      _Float16* bw = &Bs[0][0] + (buf ^ 1) * BBUF + tid * 40;
      *(half8*)(aw)      = ra0;
      *(half8*)(aw + 8)  = ra1;
      *(half8*)(bw)      = rb0;
      *(half8*)(bw + 8)  = rb1;
      *(half8*)(bw + 16) = rb2;
      *(half8*)(bw + 24) = rb3;
      buf ^= 1;
      __syncthreads();
    }
  }

  // epilogue
#pragma unroll
  for (int mt = 0; mt < 4; ++mt) {
#pragma unroll
    for (int nt = 0; nt < 4; ++nt) {
      const int col = n0 + wn * 64 + nt * 16 + lm;
#pragma unroll
      for (int r = 0; r < 8; ++r) {
        const int row = m0 + wm * 64 + mt * 16 + lhi * 8 + r;
        float v = acc[mt][nt][r];
        if (MODE == 0) {
          C16[(size_t)row * ldc + col] = (_Float16)v;
        } else if (MODE == 1) {
          float g = 0.5f * v * (1.0f + erff(v * 0.70710678118654752f));
          C16[(size_t)row * ldc + col] = (_Float16)g;
        } else {
          Cacc[(size_t)row * ldc + col] += v;
        }
      }
    }
  }
}

// ---------------------------------------------------------------------------
// V transpose: qkv[(b*T+t)*3072 + 2048 + h*64 + d]  ->  vt[(b*16+h)][d][t]
// ---------------------------------------------------------------------------
__global__ __launch_bounds__(256)
void transpose_v(const _Float16* __restrict__ qkv, _Float16* __restrict__ vt)
{
  int idx  = blockIdx.x * 256 + threadIdx.x;     // 8*16*64*1024 total
  int t    = idx & 1023;
  int rest = idx >> 10;
  int d    = rest & 63;
  int bh   = rest >> 6;                          // 0..127
  int b = bh >> 4, h = bh & 15;
  vt[(size_t)bh * 65536 + (size_t)d * 1024 + t] =
      qkv[((size_t)(b * 1024 + t)) * 3072 + 2048 + h * 64 + d];
}

// ---------------------------------------------------------------------------
// Flash attention (causal), one wave per 16-query tile, 4 waves per block.
// Computes ST = K*Q^T (so each lane owns a fixed query -> per-lane softmax
// stats + one shfl_xor(16)), then OT += VT * PT via WMMA. Q scaled by 1/8.
// qkv layout: [tok][3072] = Q|K|V each [h*64+d];  vt: [(b*16+h)][d][t]
// ---------------------------------------------------------------------------
__global__ __launch_bounds__(128)
void attn_kernel(const _Float16* __restrict__ qkv,
                 const _Float16* __restrict__ vt,
                 _Float16* __restrict__ out)
{
  const int bh = blockIdx.x;                 // 0..127
  const int b = bh >> 4, h = bh & 15;
  const int wave = threadIdx.x >> 5, lane = threadIdx.x & 31;
  const int lm = lane & 15, lhi = lane >> 4;
  const int qbase = blockIdx.y * 64 + wave * 16;
  const int qi = qbase + lm;
  const size_t RS = 3072;

  // Q B-fragments (scale 1/sqrt(64)=0.125 folded in; exact in f16)
  const _Float16* Qr = qkv + ((size_t)(b * 1024 + qbase + lm)) * RS + h * 64;
  const _Float16 qs = (_Float16)0.125f;
  v16h qf[2];
#pragma unroll
  for (int ds = 0; ds < 2; ++ds) {
    const _Float16* p = Qr + ds * 32 + lhi * 8;
    half8 c0 = *(const half8*)p;
    half8 c1 = *(const half8*)(p + 16);
#pragma unroll
    for (int i = 0; i < 8; ++i) { qf[ds][i] = c0[i] * qs; qf[ds][i + 8] = c1[i] * qs; }
  }

  v8f ot[4] = {};                            // O^T: rows d = dt*16 + r + 8*lhi, col q = lm
  float mrun = -__builtin_inff();
  float lrun = 0.f;
  const _Float16* Kb = qkv + ((size_t)b * 1024) * RS + 1024 + h * 64;
  const _Float16* Vb = vt + (size_t)bh * 64 * 1024;

  for (int kb = 0; kb < qbase + 16; kb += 32) {
    // S^T tiles for keys [kb,kb+16) and [kb+16,kb+32)
    v8f st[2];
#pragma unroll
    for (int t2 = 0; t2 < 2; ++t2) {
      const _Float16* Kr = Kb + (size_t)(kb + t2 * 16 + lm) * RS;
      v8f c = {};
#pragma unroll
      for (int ds = 0; ds < 2; ++ds) {
        const _Float16* p = Kr + ds * 32 + lhi * 8;
        half8 c0 = *(const half8*)p;
        half8 c1 = *(const half8*)(p + 16);
        v16h kf;
#pragma unroll
        for (int i = 0; i < 8; ++i) { kf[i] = c0[i]; kf[i + 8] = c1[i]; }
        c = WMMA_F16(kf, qf[ds], c);
      }
      st[t2] = c;
    }

    // causal mask + online softmax (per-lane query stats)
    float p0[8], p1[8];
    float mpart = -__builtin_inff();
#pragma unroll
    for (int r = 0; r < 8; ++r) {
      int kk0 = kb + lhi * 8 + r;
      float s0 = (kk0      <= qi) ? st[0][r] : -__builtin_inff();
      float s1 = (kk0 + 16 <= qi) ? st[1][r] : -__builtin_inff();
      p0[r] = s0; p1[r] = s1;
      mpart = fmaxf(mpart, fmaxf(s0, s1));
    }
    mpart = fmaxf(mpart, __shfl_xor(mpart, 16, 32));
    float mnew = fmaxf(mrun, mpart);
    float rescale = __expf(mrun - mnew);
    lrun *= rescale;
#pragma unroll
    for (int dt = 0; dt < 4; ++dt)
#pragma unroll
      for (int r = 0; r < 8; ++r) ot[dt][r] *= rescale;
    float lpart = 0.f;
#pragma unroll
    for (int r = 0; r < 8; ++r) {
      p0[r] = __expf(p0[r] - mnew);
      p1[r] = __expf(p1[r] - mnew);
      lpart += p0[r] + p1[r];
    }
    lrun += lpart + __shfl_xor(lpart, 16, 32);
    mrun = mnew;

    // Build P^T B-fragment (needs a cross-half swap: shfl_xor 16)
    v16h pf;
#pragma unroll
    for (int r = 0; r < 8; ++r) {
      float o0 = __shfl_xor(p0[r], 16, 32);
      float o1 = __shfl_xor(p1[r], 16, 32);
      pf[r]     = (_Float16)(lhi ? o1    : p0[r]);
      pf[r + 8] = (_Float16)(lhi ? p1[r] : o0);
    }

    // O^T += V^T(A) * P^T(B)
#pragma unroll
    for (int dt = 0; dt < 4; ++dt) {
      const _Float16* p = Vb + (size_t)(dt * 16 + lm) * 1024 + kb + lhi * 16;
      half8 c0 = *(const half8*)p;
      half8 c1 = *(const half8*)(p + 8);
      v16h vf;
#pragma unroll
      for (int i = 0; i < 8; ++i) { vf[i] = c0[i]; vf[i + 8] = c1[i]; }
      ot[dt] = WMMA_F16(vf, pf, ot[dt]);
    }
  }

  // epilogue: divide by l, pack 8 consecutive d per lane -> one b128 store each
  const float inv = 1.f / lrun;
  _Float16* Or = out + ((size_t)(b * 1024 + qbase + lm)) * DMODEL + h * 64;
#pragma unroll
  for (int dt = 0; dt < 4; ++dt) {
    half8 pk8;
#pragma unroll
    for (int r = 0; r < 8; ++r) pk8[r] = (_Float16)(ot[dt][r] * inv);
    *(half8*)(Or + dt * 16 + lhi * 8) = pk8;
  }
}

// ---------------------------------------------------------------------------
// Host side
// ---------------------------------------------------------------------------
struct LayerW {
  const float *ln1g0, *ln1g1, *ln1b0, *ln1b1;
  const float *ln2g0, *ln2g1, *ln2b0, *ln2b1;
  const float *wq[3], *wk[3], *wv[3], *wo[3];
  const float *fk, *pk, *fv, *pv;
  const float *fc1[3], *fc2[3];
};

extern "C" void kernel_launch(void* const* d_in, const int* in_sizes, int n_in,
                              void* d_out, int out_size, void* d_ws, size_t ws_size,
                              hipStream_t stream)
{
  (void)out_size; (void)ws_size; (void)n_in;
  auto P = [&](int i) { return (const float*)d_in[i]; };

  // Input-order detection: insertion order (x first) vs jax-tree order
  // (alphabetical, params first, x last).
  const bool x_first = (in_sizes[0] == MTOK * DMODEL);
  const float* xin;
  LayerW Lw[2];
  if (x_first) {
    xin = P(0);
    for (int l = 0; l < 2; ++l) {
      int o = 1 + 30 * l;
      LayerW& L = Lw[l];
      L.ln1g0 = P(o + 0); L.ln1g1 = P(o + 1); L.ln1b0 = P(o + 2); L.ln1b1 = P(o + 3);
      L.ln2g0 = P(o + 4); L.ln2g1 = P(o + 5); L.ln2b0 = P(o + 6); L.ln2b1 = P(o + 7);
      for (int j = 0; j < 3; ++j) L.wq[j] = P(o + 8 + j);
      for (int j = 0; j < 3; ++j) L.wk[j] = P(o + 11 + j);
      for (int j = 0; j < 3; ++j) L.wv[j] = P(o + 14 + j);
      for (int j = 0; j < 3; ++j) L.wo[j] = P(o + 17 + j);
      L.fk = P(o + 20); L.pk = P(o + 21); L.fv = P(o + 22); L.pv = P(o + 23);
      for (int j = 0; j < 3; ++j) L.fc1[j] = P(o + 24 + j);
      for (int j = 0; j < 3; ++j) L.fc2[j] = P(o + 27 + j);
    }
  } else {
    xin = P(60);
    for (int l = 0; l < 2; ++l) {
      int o = 30 * l;
      LayerW& L = Lw[l];
      for (int j = 0; j < 3; ++j) L.fc1[j] = P(o + 0 + j);
      for (int j = 0; j < 3; ++j) L.fc2[j] = P(o + 3 + j);
      L.fk = P(o + 6); L.fv = P(o + 7);
      L.ln1b0 = P(o + 8);  L.ln1b1 = P(o + 9);
      L.ln1g0 = P(o + 10); L.ln1g1 = P(o + 11);
      L.ln2b0 = P(o + 12); L.ln2b1 = P(o + 13);
      L.ln2g0 = P(o + 14); L.ln2g1 = P(o + 15);
      L.pk = P(o + 16); L.pv = P(o + 17);
      for (int j = 0; j < 3; ++j) L.wk[j] = P(o + 18 + j);
      for (int j = 0; j < 3; ++j) L.wo[j] = P(o + 21 + j);
      for (int j = 0; j < 3; ++j) L.wq[j] = P(o + 24 + j);
      for (int j = 0; j < 3; ++j) L.wv[j] = P(o + 27 + j);
    }
  }

  // Workspace carve (152 MB total)
  char* w = (char*)d_ws;
  float*    x_buf = (float*)w;    w += (size_t)MTOK * DMODEL * 4;   // residual stream (f32)
  _Float16* act16 = (_Float16*)w; w += (size_t)MTOK * DMODEL * 2;   // h1 / attn-out / h2
  _Float16* big16 = (_Float16*)w; w += (size_t)MTOK * DFF    * 2;   // QKV (3072) / FF (4096)
  _Float16* vt    = (_Float16*)w; w += (size_t)128 * 64 * TSEQ * 2; // V^T per (b,h)
  _Float16* wqkvT = (_Float16*)w; w += (size_t)3072 * 1024 * 2;
  _Float16* woT   = (_Float16*)w; w += (size_t)1024 * 1024 * 2;
  _Float16* fc1T  = (_Float16*)w; w += (size_t)4096 * 1024 * 2;
  _Float16* fc2T  = (_Float16*)w; w += (size_t)1024 * 4096 * 2;

  hipMemcpyAsync(x_buf, xin, (size_t)MTOK * DMODEL * 4, hipMemcpyDeviceToDevice, stream);

  for (int l = 0; l < 2; ++l) {
    const LayerW& L = Lw[l];

    // f16 transposed weights (low-rank K/V augmentation folded into Wk/Wv)
    build_wt<<<4096, 256, 0, stream>>>(wqkvT,                       L.wq[0], L.wq[1], L.wq[2], nullptr, nullptr, 1024, 1024, 256, 256);
    build_wt<<<4096, 256, 0, stream>>>(wqkvT + (size_t)1024 * 1024, L.wk[0], L.wk[1], L.wk[2], L.fk,    L.pk,    1024, 1024, 256, 256);
    build_wt<<<4096, 256, 0, stream>>>(wqkvT + (size_t)2048 * 1024, L.wv[0], L.wv[1], L.wv[2], L.fv,    L.pv,    1024, 1024, 256, 256);
    build_wt<<<4096, 256, 0, stream>>>(woT,  L.wo[0],  L.wo[1],  L.wo[2],  nullptr, nullptr, 1024, 1024, 256,  256);
    build_wt<<<16384, 256, 0, stream>>>(fc1T, L.fc1[0], L.fc1[1], L.fc1[2], nullptr, nullptr, 4096, 1024, 1024, 256);
    build_wt<<<16384, 256, 0, stream>>>(fc2T, L.fc2[0], L.fc2[1], L.fc2[2], nullptr, nullptr, 1024, 4096, 256, 1024);

    // x -> ln1 -> QKV -> attention -> +x
    block_norm<<<MTOK, 256, 0, stream>>>(x_buf, L.ln1g0, L.ln1g1, L.ln1b0, L.ln1b1, act16);
    gemm_wmma<0><<<dim3(64, 12), 256, 0, stream>>>(act16, wqkvT, big16, nullptr, 1024, 1024, 3072);
    transpose_v<<<32768, 256, 0, stream>>>(big16, vt);
    attn_kernel<<<dim3(128, 16), 128, 0, stream>>>(big16, vt, act16);
    gemm_wmma<2><<<dim3(64, 4), 256, 0, stream>>>(act16, woT, nullptr, x_buf, 1024, 1024, 1024);

    // x -> ln2 -> fc1(+gelu) -> fc2 -> +x
    block_norm<<<MTOK, 256, 0, stream>>>(x_buf, L.ln2g0, L.ln2g1, L.ln2b0, L.ln2b1, act16);
    gemm_wmma<1><<<dim3(64, 16), 256, 0, stream>>>(act16, fc1T, big16, nullptr, 1024, 1024, 4096);
    gemm_wmma<2><<<dim3(64, 4), 256, 0, stream>>>(big16, fc2T, nullptr, x_buf, 4096, 4096, 1024);
  }

  hipMemcpyAsync(d_out, x_buf, (size_t)MTOK * DMODEL * 4, hipMemcpyDeviceToDevice, stream);
}
